// LinearCausalAttention_10960756539680
// MI455X (gfx1250) — matrix operand order repaired and checked by reading, not compiled, not run on hardware
//
#include <hip/hip_runtime.h>
#include <hip/hip_bf16.h>

// ---------------------------------------------------------------------------
// CDNA5 (gfx1250) linear causal attention, bf16 WMMA (16x16x32, f32 acc).
// Phase 1: fused QKV projection GEMM   C[1536, 8*2304] = W @ X + b
// Phase 2: denominator (head-cumsum as in reference)
// Phase 3: chunked causal linear attention (chunk = 32 = WMMA K)
// ---------------------------------------------------------------------------

typedef __bf16 bh16;
typedef bh16  v16bf __attribute__((ext_vector_type(16)));
typedef float v8f   __attribute__((ext_vector_type(8)));

#define WMMA_BF16(a, b, c) \
  __builtin_amdgcn_wmma_f32_16x16x32_bf16(false, (a), false, (b), (short)0, (c), false, false)

// A-operand (16x32 bf16, M x K) loaded from a row-major [m][k] tile row.
// Layout used here: lane (tl,h): m = tl ; element e -> k = 16h + e (contiguous)
// => elements 0..15 are row[16h .. 16h+16)
__device__ __forceinline__ v16bf ld_a16(const bh16* row, int h) {
  union { v16bf v; uint4 u[2]; } r;
  r.u[0] = *(const uint4*)(row + 16 * h);
  r.u[1] = *(const uint4*)(row + 16 * h + 8);
  return r.v;
}

// B-operand (32x16 bf16, K x N) loaded from an [n][k] (transposed) tile row.
// Layout used here: lane (tl,h): n = tl ; element e -> k = 16h + e (contiguous)
__device__ __forceinline__ v16bf ld_b16(const bh16* row, int h) {
  union { v16bf v; uint4 u[2]; } r;
  r.u[0] = *(const uint4*)(row + 16 * h);
  r.u[1] = *(const uint4*)(row + 16 * h + 8);
  return r.v;
}

// branchless elu(v)+1 : v>0 ? v+1 : exp(v)   (exp arg clamped, no divergence)
__device__ __forceinline__ float elu1(float v) {
  float ev = __expf(fminf(v, 0.0f));
  return (v > 0.0f) ? (v + 1.0f) : ev;
}

// ---------------------------------------------------------------------------
// Kernel A1: pack [Wq;Wkv] -> bf16 Wb[1536][512]; also bcat[1536] = [bq;bkv]
// ---------------------------------------------------------------------------
__global__ __launch_bounds__(256) void convert_w_kernel(
    const float* __restrict__ Wq, const float* __restrict__ Wkv,
    const float* __restrict__ bq, const float* __restrict__ bkv,
    bh16* __restrict__ Wb, float* __restrict__ bcat) {
  int i = blockIdx.x * 256 + threadIdx.x;
  if (i < 1536 * 512) {
    float v = (i < 512 * 512) ? Wq[i] : Wkv[i - 512 * 512];
    Wb[i] = (bh16)v;
  }
  if (i < 1536) bcat[i] = (i < 512) ? bq[i] : bkv[i - 512];
}

// ---------------------------------------------------------------------------
// Kernel A2: x[n][c][l] f32 -> Xt[n][l][c] bf16 (tiled transpose via LDS)
// ---------------------------------------------------------------------------
__global__ __launch_bounds__(256) void transpose_x_kernel(
    const float* __restrict__ x, bh16* __restrict__ Xt) {
  __shared__ float tile[32][33];
  const int n  = blockIdx.z;
  const int l0 = blockIdx.x * 32;
  const int c0 = blockIdx.y * 32;
  const int tx = threadIdx.x, ty = threadIdx.y;
#pragma unroll
  for (int rep = 0; rep < 4; ++rep) {
    int c = ty + rep * 8;
    tile[c][tx] = x[((size_t)n * 512 + c0 + c) * 2304 + l0 + tx];
  }
  __syncthreads();
#pragma unroll
  for (int rep = 0; rep < 4; ++rep) {
    int l = ty + rep * 8;
    Xt[((size_t)n * 2304 + l0 + l) * 512 + c0 + tx] = (bh16)tile[tx][l];
  }
}

// ---------------------------------------------------------------------------
// Kernel B: projection GEMM, 128x128 tile per block (8 waves, 2x4), K-step 32.
//   Epilogue: +bias, elu+1 on q/K, scatter:
//     Qf[n][h][l][d]  (l-major, A-operand friendly)
//     Kd[n][h][d][l]  (d-major: B-operand for Q@K^T + A-operand for K^T@V)
//     Vd[n][h][d][l]  (d-major: B-operand for P@V and K^T@V)
// ---------------------------------------------------------------------------
__global__ __launch_bounds__(256) void gemm_proj_kernel(
    const bh16* __restrict__ Wb, const bh16* __restrict__ Xt,
    const float* __restrict__ bcat,
    bh16* __restrict__ Qf, bh16* __restrict__ Kd, bh16* __restrict__ Vd) {
  __shared__ __attribute__((aligned(16))) bh16 Asm[128][32];  // W tile [o][c]
  __shared__ __attribute__((aligned(16))) bh16 Bsm[128][32];  // X tile [l][c]

  const int tid   = threadIdx.x;
  const int nb    = blockIdx.z;
  const int oBase = blockIdx.x * 128;
  const int lBase = blockIdx.y * 128;
  const int lane = tid & 31, wv = tid >> 5;
  const int tl = lane & 15, hh = lane >> 4;
  const int wm = wv >> 2, wn = wv & 3;  // wave grid 2 x 4 over (64 x 32) each

  v8f acc[4][2] = {};

  const int arow = tid >> 1, ahalf = tid & 1;
  const size_t wbase = (size_t)(oBase + arow) * 512 + 16 * ahalf;
  const size_t xbase = ((size_t)nb * 2304 + lBase + arow) * 512 + 16 * ahalf;

  for (int kb = 0; kb < 16; ++kb) {
    const bh16* wsrc = Wb + wbase + (size_t)kb * 32;
    const bh16* xsrc = Xt + xbase + (size_t)kb * 32;
    *(uint4*)&Asm[arow][16 * ahalf]     = *(const uint4*)(wsrc);
    *(uint4*)&Asm[arow][16 * ahalf + 8] = *(const uint4*)(wsrc + 8);
    *(uint4*)&Bsm[arow][16 * ahalf]     = *(const uint4*)(xsrc);
    *(uint4*)&Bsm[arow][16 * ahalf + 8] = *(const uint4*)(xsrc + 8);
    __syncthreads();

    v16bf av[4], bv[2];
#pragma unroll
    for (int mi = 0; mi < 4; ++mi)
      av[mi] = ld_a16(&Asm[wm * 64 + mi * 16 + tl][0], hh);
#pragma unroll
    for (int ni = 0; ni < 2; ++ni)
      bv[ni] = ld_b16(&Bsm[wn * 32 + ni * 16 + tl][0], hh);
#pragma unroll
    for (int mi = 0; mi < 4; ++mi)
#pragma unroll
      for (int ni = 0; ni < 2; ++ni)
        acc[mi][ni] = WMMA_BF16(av[mi], bv[ni], acc[mi][ni]);
    __syncthreads();
  }

  // Epilogue. C layout: m = r + 8*hh, n = tl.
#pragma unroll
  for (int mi = 0; mi < 4; ++mi) {
    // biases for this tile's 8 rows (contiguous in r): two vector loads
    const int obase = oBase + wm * 64 + mi * 16 + 8 * hh;
    const float4 bb0 = *(const float4*)&bcat[obase];
    const float4 bb1 = *(const float4*)&bcat[obase + 4];
#pragma unroll
    for (int ni = 0; ni < 2; ++ni) {
#pragma unroll
      for (int r = 0; r < 8; ++r) {
        int o = obase + r;
        int l = lBase + wn * 32 + ni * 16 + tl;
        float v = acc[mi][ni][r] + ((r < 4) ? bb0[r] : bb1[r - 4]);
        if (o < 1024) v = elu1(v);  // block-uniform branch (oBase % 128 == 0)
        if (o < 512) {
          int hd = o >> 6, d = o & 63;
          Qf[(((size_t)nb * 8 + hd) * 2304 + l) * 64 + d] = (bh16)v;
        } else if (o < 1024) {
          int o2 = o - 512; int hd = o2 >> 6, d = o2 & 63;
          Kd[(((size_t)nb * 8 + hd) * 64 + d) * 2304 + l] = (bh16)v;
        } else {
          int o3 = o - 1024; int hd = o3 >> 6, d = o3 & 63;
          Vd[(((size_t)nb * 8 + hd) * 64 + d) * 2304 + l] = (bh16)v;
        }
      }
    }
  }
}

// ---------------------------------------------------------------------------
// Kernel C: denominator, replicating reference cumsum over the HEAD axis:
//   den[n,h,l] = 1 / (sum_d Q[n,h,l,d] * sum_{h'<=h} K[n,h',l,d] + 1e-10)
// ---------------------------------------------------------------------------
__global__ __launch_bounds__(256) void denom_kernel(
    const bh16* __restrict__ Qf, const bh16* __restrict__ Kd,
    float* __restrict__ den) {
  int idx = blockIdx.x * 256 + threadIdx.x;  // (nb, l)
  if (idx >= 8 * 2304) return;
  int nb = idx / 2304, l = idx - nb * 2304;
  float ks[64];
#pragma unroll
  for (int d = 0; d < 64; ++d) ks[d] = 0.0f;
  for (int hd = 0; hd < 8; ++hd) {
    const bh16* kp = Kd + (((size_t)nb * 8 + hd) * 64) * 2304 + l;
    const bh16* qp = Qf + (((size_t)nb * 8 + hd) * 2304 + l) * 64;
    float dot = 0.0f;
#pragma unroll
    for (int d = 0; d < 64; ++d) {
      ks[d] += (float)kp[(size_t)d * 2304];
      dot += (float)qp[d] * ks[d];
    }
    den[((size_t)nb * 8 + hd) * 2304 + l] = 1.0f / (dot + 1e-10f);
  }
}

// ---------------------------------------------------------------------------
// Kernel D: chunked causal linear attention. 1 block per (n,head), 4 waves.
// Chunk C=32 (== WMMA K). State S[64][64] f32 lives in VGPRs: wave w owns
// dv-columns [16w,16w+16) as 4 C-tiles. Per chunk:
//   P = causal_mask(Q_t K_t^T)          (each wave one 16x16 C-tile, 2 WMMA)
//   O = P V_t + Q_t S_prev              (2 + 4 WMMA per wave)
//   S += K_t^T V_t                      (4 WMMA per wave, f32 acc in place)
//   out = O * den
// ---------------------------------------------------------------------------
__global__ __launch_bounds__(128) void attn_kernel(
    const bh16* __restrict__ Qf, const bh16* __restrict__ Kd,
    const bh16* __restrict__ Vd, const float* __restrict__ den,
    float* __restrict__ out) {
  __shared__ __attribute__((aligned(16))) bh16 Qs[32][64];  // [l][d]
  __shared__ __attribute__((aligned(16))) bh16 Ks[64][32];  // [d][l]
  __shared__ __attribute__((aligned(16))) bh16 Vs[64][32];  // [d][l]
  __shared__ __attribute__((aligned(16))) bh16 Ps[32][32];  // [l][l'] masked
  __shared__ __attribute__((aligned(16))) bh16 Ss[64][64];  // [dv][dk] (B layout)

  const int seq = blockIdx.x;          // nb*8 + head
  const int nb = seq >> 3, hd = seq & 7;
  const int tid = threadIdx.x;
  const int lane = tid & 31, w = tid >> 5;
  const int tl = lane & 15, hh = lane >> 4;
  const int pa = w >> 1, pip = w & 1;  // this wave's P tile (row-half, col-half)

  const bh16* Qg = Qf + (size_t)seq * 2304 * 64;
  const bh16* Kg = Kd + (size_t)seq * 64 * 2304;
  const bh16* Vg = Vd + (size_t)seq * 64 * 2304;
  const float* dg = den + (size_t)seq * 2304;
  float* og = out + ((size_t)nb * 512 + (size_t)hd * 64) * 2304;

  v8f S[4] = {};  // S[dk-tile i][.] : columns dv in [16w, 16w+16)

  const int krow = tid >> 1, khalf = tid & 1;

  for (int t = 0; t < 72; ++t) {
    const int l0 = t * 32;
    __syncthreads();  // previous chunk's LDS reads complete

    // cooperative loads (all flat-contiguous, 32B per thread per tensor)
    {
      const bh16* qsrc = Qg + (size_t)l0 * 64 + (size_t)tid * 16;
      bh16* qdst = &Qs[0][0] + tid * 16;
      *(uint4*)qdst       = *(const uint4*)qsrc;
      *(uint4*)(qdst + 8) = *(const uint4*)(qsrc + 8);
      const bh16* ksrc = Kg + (size_t)krow * 2304 + l0 + 16 * khalf;
      *(uint4*)&Ks[krow][16 * khalf]     = *(const uint4*)ksrc;
      *(uint4*)&Ks[krow][16 * khalf + 8] = *(const uint4*)(ksrc + 8);
      const bh16* vsrc = Vg + (size_t)krow * 2304 + l0 + 16 * khalf;
      *(uint4*)&Vs[krow][16 * khalf]     = *(const uint4*)vsrc;
      *(uint4*)&Vs[krow][16 * khalf + 8] = *(const uint4*)(vsrc + 8);
    }
    // stage S (pre-update state) into LDS as bf16 in B-operand layout [dv][dk]
#pragma unroll
    for (int i = 0; i < 4; ++i) {
      union { bh16 b[8]; uint4 u; } pk;
#pragma unroll
      for (int r = 0; r < 8; ++r) pk.b[r] = (bh16)S[i][r];
      *(uint4*)&Ss[16 * w + tl][16 * i + 8 * hh] = pk.u;
    }
    __syncthreads();

    // P tile (pa, pip):  P[l][l'] = sum_d Q[l][d] * K[l'][d]
    {
      v8f pc = {};
#pragma unroll
      for (int kk = 0; kk < 2; ++kk) {
        v16bf aq = ld_a16(&Qs[16 * pa + tl][32 * kk], hh);
        union { v16bf v; bh16 b[16]; } bk;  // B[k=d][n=l'] gather from [d][l]
#pragma unroll
        for (int e = 0; e < 16; ++e)
          bk.b[e] = Ks[32 * kk + 16 * hh + e][16 * pip + tl];
        pc = WMMA_BF16(aq, bk.v, pc);
      }
      const int lc = 16 * pip + tl;
#pragma unroll
      for (int r = 0; r < 8; ++r) {
        int lr = 16 * pa + r + 8 * hh;
        Ps[lr][lc] = (bh16)((lc <= lr) ? pc[r] : 0.0f);  // causal (diag incl.)
      }
    }
    __syncthreads();

    // O = P @ V + Q @ S_prev ;  S += K^T @ V
    v8f oc[2] = {};
    v16bf bv = ld_b16(&Vs[16 * w + tl][0], hh);  // B[k=l'][n=dv]
#pragma unroll
    for (int a2 = 0; a2 < 2; ++a2) {
      v16bf ap = ld_a16(&Ps[16 * a2 + tl][0], hh);       // A[m=l][k=l']
      oc[a2] = WMMA_BF16(ap, bv, oc[a2]);
#pragma unroll
      for (int kk = 0; kk < 2; ++kk) {
        v16bf aq = ld_a16(&Qs[16 * a2 + tl][32 * kk], hh);   // A[m=l][k=dk]
        v16bf bs = ld_b16(&Ss[16 * w + tl][32 * kk], hh);    // B[k=dk][n=dv]
        oc[a2] = WMMA_BF16(aq, bs, oc[a2]);
      }
    }
#pragma unroll
    for (int i = 0; i < 4; ++i) {
      v16bf ak = ld_a16(&Ks[16 * i + tl][0], hh);  // A[m=dk][k=l] = K^T
      S[i] = WMMA_BF16(ak, bv, S[i]);
    }

    // write out[n][64*head + dv][l] = O * den
    const int dv = 16 * w + tl;
#pragma unroll
    for (int a2 = 0; a2 < 2; ++a2) {
#pragma unroll
      for (int r = 0; r < 8; ++r) {
        int l = l0 + 16 * a2 + r + 8 * hh;
        og[(size_t)dv * 2304 + l] = oc[a2][r] * dg[l];
      }
    }
  }
}

// ---------------------------------------------------------------------------
// Launch. Workspace usage: ~77.7 MB
//   Wb 1.5MB | Xt 18.9MB | Qf 18.9MB | Kd 18.9MB | Vd 18.9MB | den+bias 0.6MB
// ---------------------------------------------------------------------------
extern "C" void kernel_launch(void* const* d_in, const int* in_sizes, int n_in,
                              void* d_out, int out_size, void* d_ws, size_t ws_size,
                              hipStream_t stream) {
  (void)in_sizes; (void)n_in; (void)out_size; (void)ws_size;
  const float* x   = (const float*)d_in[0];
  const float* Wq  = (const float*)d_in[1];
  const float* bq  = (const float*)d_in[2];
  const float* Wkv = (const float*)d_in[3];
  const float* bkv = (const float*)d_in[4];
  float* out = (float*)d_out;

  char* ws = (char*)d_ws;
  size_t off = 0;
  auto carve = [&](size_t bytes) -> void* {
    void* p = ws + off;
    off += (bytes + 255) & ~(size_t)255;
    return p;
  };
  bh16* Wb   = (bh16*)carve((size_t)1536 * 512 * 2);
  bh16* Xt   = (bh16*)carve((size_t)8 * 2304 * 512 * 2);
  bh16* Qf   = (bh16*)carve((size_t)8 * 512 * 2304 * 2);
  bh16* Kd   = (bh16*)carve((size_t)8 * 512 * 2304 * 2);
  bh16* Vd   = (bh16*)carve((size_t)8 * 512 * 2304 * 2);
  float* dn  = (float*)carve((size_t)8 * 8 * 2304 * 4);
  float* bct = (float*)carve((size_t)1536 * 4);

  convert_w_kernel<<<(1536 * 512 + 255) / 256, 256, 0, stream>>>(Wq, Wkv, bq, bkv, Wb, bct);
  transpose_x_kernel<<<dim3(72, 16, 8), dim3(32, 8), 0, stream>>>(x, Xt);
  gemm_proj_kernel<<<dim3(12, 18, 8), 256, 0, stream>>>(Wb, Xt, bct, Qf, Kd, Vd);
  denom_kernel<<<(8 * 2304 + 255) / 256, 256, 0, stream>>>(Qf, Kd, dn);
  attn_kernel<<<64, 128, 0, stream>>>(Qf, Kd, Vd, dn, out);
}
